// PointViG_82171314307287
// MI455X (gfx1250) — compile-verified, matrix operand order checked by reading, not run on hardware
//
#include <hip/hip_runtime.h>
#include <hip/hip_bf16.h>
#include <stdint.h>

typedef __attribute__((ext_vector_type(16))) __bf16        v16bf;
typedef __attribute__((ext_vector_type(8)))  float         v8f;
typedef __attribute__((ext_vector_type(8)))  unsigned int  v8u;

#define NN 50000
#define NE 800000
#define MW 2   // row tiles (16 rows each) per wave

__device__ __forceinline__ unsigned short f2bf(float f){
  unsigned u = __float_as_uint(f);
  unsigned r = u + 0x7FFFu + ((u >> 16) & 1u);
  return (unsigned short)(r >> 16);
}
__device__ __forceinline__ float bf2f(unsigned short h){
  return __uint_as_float(((unsigned)h) << 16);
}
__device__ __forceinline__ int koff(int j){ return (j < 4) ? (j << 1) : (16 + ((j - 4) << 1)); }
__device__ __forceinline__ unsigned fkey(float f){
  unsigned u = __float_as_uint(f);
  return (u & 0x80000000u) ? ~u : (u | 0x80000000u);
}
#define KEY_NEG_INF 0x007FFFFFu   // fkey(-inf)

// ---- pack fp32 weight [K,Fo] row-major -> WMMA bf16 B-fragment-major dwords ----
// layout: dword index = ((ct*nkb + kk)*32 + lane)*8 + j ; zero-padded to ceil32(K) x ceil16(Fo)
__global__ void pack_w_kernel(const float* __restrict__ w, unsigned int* __restrict__ dst,
                              int K, int Fo, int nkb, int nct){
  int tid = blockIdx.x * blockDim.x + threadIdx.x;
  int total = nct * nkb * 256;
  if (tid >= total) return;
  int j = tid & 7, lane = (tid >> 3) & 31, g = tid >> 8;
  int kk = g % nkb, ct = g / nkb;
  int half = lane >> 4, n = ct * 16 + (lane & 15);
  int k0 = kk * 32 + koff(j) + 8 * half;
  float a0 = (k0     < K && n < Fo) ? w[(size_t)k0 * Fo + n]       : 0.f;
  float a1 = (k0 + 1 < K && n < Fo) ? w[(size_t)(k0 + 1) * Fo + n] : 0.f;
  dst[tid] = ((unsigned)f2bf(a0)) | (((unsigned)f2bf(a1)) << 16);
}

// ---- x [N,6] f32 -> [N,8] bf16 (zero pad cols 6..7) ----
__global__ void pack_x_kernel(const float* __restrict__ x, unsigned short* __restrict__ xb, int n){
  int tid = blockIdx.x * blockDim.x + threadIdx.x;
  if (tid >= n * 8) return;
  int r = tid >> 3, c = tid & 7;
  xb[tid] = f2bf((c < 6) ? x[r * 6 + c] : 0.f);
}

// ---- generic fused GEMM: out = act(A(bf16)*W(bf16) + bias + pre) + post ----
// A reads are unguarded: K-padding is handled by zero-padded weights, row overrun by
// clamping the row pointer (garbage rows are never stored). All lda are multiples of 8.
__global__ __launch_bounds__(256) void gemm_kernel(
    const unsigned short* __restrict__ A, int lda, int nkb,
    const unsigned int* __restrict__ Wpk,
    const float* __restrict__ bias,
    unsigned short* __restrict__ outB, float* __restrict__ outF, int ldOut, int Fo,
    const float* __restrict__ preF, int ldPre,
    const unsigned short* __restrict__ postB, int ldPost,
    int act, int Nrows){
  int lane = threadIdx.x & 31;
  int wave = threadIdx.x >> 5;
  int mBase = (blockIdx.x * 8 + wave) * (16 * MW);
  if (mBase >= Nrows) return;                 // wave-uniform
  int half = lane >> 4, nl = lane & 15;

  const unsigned short* Ar[MW];
#pragma unroll
  for (int t = 0; t < MW; ++t){
    int m = mBase + t * 16 + nl;
    if (m > Nrows - 1) m = Nrows - 1;         // clamp: safe, rows masked at store
    Ar[t] = A + (size_t)m * lda + 8 * half;
  }

  v8f c[MW];
#pragma unroll
  for (int t = 0; t < MW; ++t) c[t] = (v8f){0.f,0.f,0.f,0.f,0.f,0.f,0.f,0.f};

  const unsigned int* wp = Wpk + ((size_t)((size_t)blockIdx.y * nkb) * 32 + lane) * 8;
  for (int kk = 0; kk < nkb; ++kk){
    uint4 b0 = ((const uint4*)wp)[0];
    uint4 b1 = ((const uint4*)wp)[1];
    wp += 256;
    v8u bu;
    bu[0]=b0.x; bu[1]=b0.y; bu[2]=b0.z; bu[3]=b0.w;
    bu[4]=b1.x; bu[5]=b1.y; bu[6]=b1.z; bu[7]=b1.w;
    v16bf bf = __builtin_bit_cast(v16bf, bu);
    int kb = kk * 32;
#pragma unroll
    for (int t = 0; t < MW; ++t){
      uint4 a0 = *(const uint4*)(Ar[t] + kb);        // k offsets {0,2,4,6}+8*half
      uint4 a1 = *(const uint4*)(Ar[t] + kb + 16);   // k offsets {16,18,20,22}+8*half
      v8u au;
      au[0]=a0.x; au[1]=a0.y; au[2]=a0.z; au[3]=a0.w;
      au[4]=a1.x; au[5]=a1.y; au[6]=a1.z; au[7]=a1.w;
      c[t] = __builtin_amdgcn_wmma_f32_16x16x32_bf16(false, __builtin_bit_cast(v16bf, au),
                                                     false, bf, (short)0, c[t], false, false);
    }
  }

  int n = blockIdx.y * 16 + nl;
  if (n >= Fo) return;                        // divergent ok post-WMMA
  float bn = bias ? bias[n] : 0.f;
#pragma unroll
  for (int t = 0; t < MW; ++t){
#pragma unroll
    for (int j = 0; j < 8; ++j){
      int mm = mBase + t * 16 + j + (half << 3);
      if (mm >= Nrows) continue;
      float y = c[t][j] + bn;
      if (preF)  y += preF[(size_t)mm * ldPre + n];
      if (act == 1)      y = fmaxf(y, 0.f);
      else if (act == 2) y = tanhf(y);
      if (postB) y += bf2f(postB[(size_t)mm * ldPost + n]);
      if (outB)  outB[(size_t)mm * ldOut + n] = f2bf(y);
      if (outF)  outF[(size_t)mm * ldOut + n] = y;
    }
  }
}

// ---- segment-max machinery ----
__global__ void fill_kernel(unsigned int* __restrict__ p, unsigned int v, int n){
  int i = blockIdx.x * blockDim.x + threadIdx.x;
  if (i < n) p[i] = v;
}
__global__ void edge_kernel(const int* __restrict__ ei, const unsigned short* __restrict__ fi,
                            unsigned int* __restrict__ agg, int F, int cshift){
  int gid = blockIdx.x * blockDim.x + threadIdx.x;
  int total = NE << cshift;
  if (gid >= total) return;
  int e = gid >> cshift;
  int f = (gid & ((1 << cshift) - 1)) << 3;
  int s = ei[e], d = ei[NE + e];
  uint4 vd = *(const uint4*)(fi + (size_t)d * F + f);
  uint4 vs = *(const uint4*)(fi + (size_t)s * F + f);
  unsigned dw[4] = {vd.x, vd.y, vd.z, vd.w};
  unsigned sw[4] = {vs.x, vs.y, vs.z, vs.w};
  unsigned int* base = agg + (size_t)d * F + f;
#pragma unroll
  for (int i = 0; i < 4; ++i){
    float d0 = bf2f((unsigned short)(dw[i] & 0xFFFFu)) - bf2f((unsigned short)(sw[i] & 0xFFFFu));
    float d1 = bf2f((unsigned short)(dw[i] >> 16))     - bf2f((unsigned short)(sw[i] >> 16));
    atomicMax(base + 2 * i,     fkey(d0));
    atomicMax(base + 2 * i + 1, fkey(d1));
  }
}
__global__ void fixup_kernel(const unsigned int* __restrict__ agg,
                             unsigned short* __restrict__ aggf, int n){
  int i = blockIdx.x * blockDim.x + threadIdx.x;
  if (i >= n) return;
  unsigned k = agg[i];
  float v = 0.f;
  if (k != KEY_NEG_INF){
    unsigned b = (k & 0x80000000u) ? (k ^ 0x80000000u) : ~k;
    v = __uint_as_float(b);
    if (((b >> 23) & 0xFFu) == 0xFFu) v = 0.f;  // isfinite guard
  }
  aggf[i] = f2bf(v);
}

extern "C" void kernel_launch(void* const* d_in, const int* in_sizes, int n_in,
                              void* d_out, int out_size, void* d_ws, size_t ws_size,
                              hipStream_t stream){
  (void)in_sizes; (void)n_in; (void)out_size; (void)ws_size;
  const float* x = (const float*)d_in[0];
  const int* ei  = (const int*)d_in[1];
  int p = 2;
  struct Lin { const float* w; const float* b; int K, Fo; };
  auto lin = [&](int K, int Fo){
    Lin L{(const float*)d_in[p], (const float*)d_in[p + 1], K, Fo}; p += 2; return L;
  };
  Lin emb = lin(6, 64);
  const int DI[3] = {64, 64, 128};
  const int FF[3] = {64, 128, 256};
  struct PG { Lin m1[3], fc1, m2[3], fc2, m3[3]; } pg[3];
  for (int l = 0; l < 3; ++l){
    int di = DI[l], F = FF[l];
    pg[l].m1[0] = lin(di, 64); pg[l].m1[1] = lin(64, 128); pg[l].m1[2] = lin(128, F);
    pg[l].fc1   = lin(F, F);
    pg[l].m2[0] = lin(F, 32);  pg[l].m2[1] = lin(32, 64);  pg[l].m2[2] = lin(64, F);
    pg[l].fc2   = lin(2 * F, F);
    pg[l].m3[0] = lin(F, 128); pg[l].m3[1] = lin(128, 64); pg[l].m3[2] = lin(64, F);
  }
  Lin hd0 = lin(256, 128), hd1 = lin(128, 64), hd2 = lin(64, 40);

  // ---- workspace arena (each alloc gets 512B slack: unguarded A reads may overrun) ----
  const int N = NN;
  char* ws = (char*)d_ws;
  size_t off = 0;
  auto alloc = [&](size_t bytes){
    size_t o = off; off = (off + bytes + 512 + 255) & ~(size_t)255; return o;
  };
  size_t O_XB  = alloc((size_t)N * 8   * 2);
  size_t O_H   = alloc((size_t)N * 256 * 2);
  size_t O_F1  = alloc((size_t)N * 256 * 2);
  size_t O_F2  = alloc((size_t)N * 256 * 2);
  size_t O_AGF = alloc((size_t)N * 256 * 2);
  size_t O_S1  = alloc((size_t)N * 128 * 2);
  size_t O_S2  = alloc((size_t)N * 128 * 2);
  size_t O_AGU = alloc((size_t)N * 256 * 4);   // u32 keys; reused as f32 PRE
  size_t O_WPK = alloc((size_t)4 * 1024 * 1024);
  unsigned short* XB  = (unsigned short*)(ws + O_XB);
  unsigned short* H   = (unsigned short*)(ws + O_H);
  unsigned short* F1  = (unsigned short*)(ws + O_F1);
  unsigned short* F2  = (unsigned short*)(ws + O_F2);
  unsigned short* AGF = (unsigned short*)(ws + O_AGF);
  unsigned short* S1  = (unsigned short*)(ws + O_S1);
  unsigned short* S2  = (unsigned short*)(ws + O_S2);
  unsigned int*   AGU = (unsigned int*)(ws + O_AGU);
  float*          PRE = (float*)(ws + O_AGU);

  // ---- weight packing ----
  unsigned int* wpkBase = (unsigned int*)(ws + O_WPK);
  size_t wcur = 0;
  struct PW { const unsigned int* ptr; int nkb, nct; };
  auto packW = [&](const float* w, int K, int Fo)->PW{
    int nkb = (K + 31) / 32, nct = (Fo + 15) / 16;
    unsigned int* dst = wpkBase + wcur;
    int total = nct * nkb * 256;
    pack_w_kernel<<<(total + 255) / 256, 256, 0, stream>>>(w, dst, K, Fo, nkb, nct);
    wcur += (size_t)total;
    return PW{dst, nkb, nct};
  };
  PW Pemb = packW(emb.w, 6, 64);
  PW Pm1[3][3], Pfc1[3], Pm2[3][3], Pfc2t[3], Pfc2b[3], Pm3[3][3];
  for (int l = 0; l < 3; ++l){
    int F = FF[l];
    for (int i = 0; i < 3; ++i) Pm1[l][i] = packW(pg[l].m1[i].w, pg[l].m1[i].K, pg[l].m1[i].Fo);
    Pfc1[l] = packW(pg[l].fc1.w, F, F);
    for (int i = 0; i < 3; ++i) Pm2[l][i] = packW(pg[l].m2[i].w, pg[l].m2[i].K, pg[l].m2[i].Fo);
    Pfc2t[l] = packW(pg[l].fc2.w,                  F, F);  // rows 0..F-1  (× fi)
    Pfc2b[l] = packW(pg[l].fc2.w + (size_t)F * F,  F, F);  // rows F..2F-1 (× a)
    for (int i = 0; i < 3; ++i) Pm3[l][i] = packW(pg[l].m3[i].w, pg[l].m3[i].K, pg[l].m3[i].Fo);
  }
  PW Ph0 = packW(hd0.w, 256, 128), Ph1 = packW(hd1.w, 128, 64), Ph2 = packW(hd2.w, 64, 40);

  pack_x_kernel<<<(N * 8 + 255) / 256, 256, 0, stream>>>(x, XB, N);

  auto gemm = [&](const unsigned short* A, int lda, PW W, const float* bias,
                  unsigned short* outB, float* outF, int ldOut, int Fo,
                  const float* pre, int ldPre, const unsigned short* post, int ldPost, int act){
    dim3 grid((N + 128 * MW - 1) / (128 * MW), W.nct);
    gemm_kernel<<<grid, 256, 0, stream>>>(A, lda, W.nkb, W.ptr, bias,
                                          outB, outF, ldOut, Fo, pre, ldPre, post, ldPost, act, N);
  };

  // ---- forward ----
  gemm(XB, 8, Pemb, emb.b, H, nullptr, 64, 64, nullptr, 0, nullptr, 0, 0);      // h = emb(x)

  for (int l = 0; l < 3; ++l){
    int di = DI[l], F = FF[l];
    gemm(H,  di,  Pm1[l][0], pg[l].m1[0].b, S1, nullptr,  64,  64, nullptr,0, nullptr,0, 1);
    gemm(S1, 64,  Pm1[l][1], pg[l].m1[1].b, S2, nullptr, 128, 128, nullptr,0, nullptr,0, 1);
    gemm(S2, 128, Pm1[l][2], pg[l].m1[2].b, F1, nullptr,   F,   F, nullptr,0,
         (l == 0) ? H : nullptr, di, 0);                                        // f1 (+h if skip)
    gemm(F1, F,   Pfc1[l],   pg[l].fc1.b,   H,  nullptr,   F,   F, nullptr,0, nullptr,0, 1); // fi -> H
    // segment_max(fi[dst]-fi[src])
    int nAgg = N * F;
    fill_kernel<<<(nAgg + 255) / 256, 256, 0, stream>>>(AGU, KEY_NEG_INF, nAgg);
    int cshift = (F == 64) ? 3 : (F == 128) ? 4 : 5;
    int etotal = NE << cshift;
    edge_kernel<<<(etotal + 255) / 256, 256, 0, stream>>>(ei, H, AGU, F, cshift);
    fixup_kernel<<<(nAgg + 255) / 256, 256, 0, stream>>>(AGU, AGF, nAgg);
    // mlp2
    gemm(AGF, F,  Pm2[l][0], pg[l].m2[0].b, S1,  nullptr, 32, 32, nullptr,0, nullptr,0, 2);
    gemm(S1, 32,  Pm2[l][1], pg[l].m2[1].b, S2,  nullptr, 64, 64, nullptr,0, nullptr,0, 2);
    gemm(S2, 64,  Pm2[l][2], pg[l].m2[2].b, AGF, nullptr,  F,  F, nullptr,0, nullptr,0, 0); // a -> AGF
    // fc2 over concat(fi,a): two accumulating GEMMs (no concat buffer)
    gemm(H,   F,  Pfc2t[l],  nullptr,       nullptr, PRE,  F,  F, nullptr,0, nullptr,0, 0);
    gemm(AGF, F,  Pfc2b[l],  pg[l].fc2.b,   F2, nullptr,   F,  F, PRE, F, F1, F, 1);        // f2
    // mlp3 + residual
    gemm(F2, F,   Pm3[l][0], pg[l].m3[0].b, S1, nullptr, 128, 128, nullptr,0, nullptr,0, 1);
    gemm(S1, 128, Pm3[l][1], pg[l].m3[1].b, S2, nullptr,  64,  64, nullptr,0, nullptr,0, 1);
    gemm(S2, 64,  Pm3[l][2], pg[l].m3[2].b, H,  nullptr,   F,   F, nullptr,0, F2, F, 0);    // h -> H
  }
  // head
  gemm(H,  256, Ph0, hd0.b, S1, nullptr, 128, 128, nullptr,0, nullptr,0, 2);
  gemm(S1, 128, Ph1, hd1.b, S2, nullptr,  64,  64, nullptr,0, nullptr,0, 1);
  gemm(S2,  64, Ph2, hd2.b, nullptr, (float*)d_out, 40, 40, nullptr,0, nullptr,0, 0);
}
